// SRUCell_53592601919991
// MI455X (gfx1250) — matrix operand order repaired and checked by reading, not compiled, not run on hardware
//
#include <hip/hip_runtime.h>
#include <hip/hip_bf16.h>

typedef __attribute__((ext_vector_type(16))) __bf16 v16bf;
typedef __attribute__((ext_vector_type(8)))  float  v8f;
typedef int v4i __attribute__((vector_size(16)));

#define L_TOTAL 2048
#define BATCH   16
#define DDIM    1024
#define KDIM    1024
#define NDIM    3072
#define LCHUNK  256
#define MCHUNK  (LCHUNK * BATCH)   // 4096 GEMM rows per chunk

#define BM 128
#define BN 128
#define TK 32
#define ASTR 40   // u16 units per LDS row (80 B: 16B-aligned, padded to dodge bank conflicts)

#if defined(__gfx1250__) && __has_builtin(__builtin_amdgcn_global_load_async_to_lds_b128)
#define HAVE_ASYNC_LDS 1
#endif

static __device__ __forceinline__ unsigned short f2bf(float f) {
    union { float f; unsigned u; } v; v.f = f;
    unsigned u = v.u + 0x7FFFu + ((v.u >> 16) & 1u);   // round-to-nearest-even
    return (unsigned short)(u >> 16);
}
static __device__ __forceinline__ unsigned pack2(float a, float b) {
    return (unsigned)f2bf(a) | ((unsigned)f2bf(b) << 16);
}

union Frag { v16bf v; uint4 q[2]; };

// 16-byte global(bf16) -> LDS copy: async on CDNA5, sync fallback elsewhere.
static __device__ __forceinline__ void cp16(unsigned short* lds_dst,
                                            const unsigned short* gsrc) {
#if defined(HAVE_ASYNC_LDS)
    __builtin_amdgcn_global_load_async_to_lds_b128(
        (__attribute__((address_space(1))) v4i*)(gsrc),
        (__attribute__((address_space(3))) v4i*)(lds_dst), 0, 0);
#else
    *(uint4*)lds_dst = *(const uint4*)gsrc;
#endif
}
static __device__ __forceinline__ void async_wait() {
#if defined(HAVE_ASYNC_LDS)
#if __has_builtin(__builtin_amdgcn_s_wait_asynccnt)
    __builtin_amdgcn_s_wait_asynccnt(0);
#else
    asm volatile("s_wait_asynccnt 0" ::: "memory");
#endif
#endif
}

// ---------------------------------------------------------------------------
// One-time: W (K x N fp32) -> Wt (N x K bf16), LDS 32x32 tile transpose.
// ---------------------------------------------------------------------------
__global__ __launch_bounds__(256) void sru_wt(const float* __restrict__ W,
                                              unsigned short* __restrict__ Wt) {
    __shared__ float tile[32][33];
    const int tx = threadIdx.x & 31;
    const int ty = threadIdx.x >> 5;           // 0..7
    const int n0 = blockIdx.x * 32;
    const int k0 = blockIdx.y * 32;
#pragma unroll
    for (int i = 0; i < 4; ++i)
        tile[ty + i * 8][tx] = W[(size_t)(k0 + ty + i * 8) * NDIM + n0 + tx];
    __syncthreads();
#pragma unroll
    for (int i = 0; i < 4; ++i)
        Wt[(size_t)(n0 + ty + i * 8) * KDIM + k0 + tx] = f2bf(tile[tx][ty + i * 8]);
}

// ---------------------------------------------------------------------------
// Per chunk: x slab fp32 -> bf16 (same layout), 8 elements/thread vectorized.
// ---------------------------------------------------------------------------
__global__ __launch_bounds__(256) void sru_xb(const float* __restrict__ X,
                                              unsigned short* __restrict__ Xb) {
    const size_t idx = ((size_t)blockIdx.x * blockDim.x + threadIdx.x) * 8;
    const float4* s = (const float4*)(X + idx);
    float4 a = s[0], b = s[1];
    uint4 p;
    p.x = pack2(a.x, a.y); p.y = pack2(a.z, a.w);
    p.z = pack2(b.x, b.y); p.w = pack2(b.z, b.w);
    *(uint4*)(Xb + idx) = p;
}

// ---------------------------------------------------------------------------
// GEMM: U = Xb @ Wt^T  (bf16 WMMA, fp32 accumulate)
// block 128x128, 8 waves, wave tile 64x32 (4x2 WMMA tiles), TK=32.
// Double-buffered LDS: async global->LDS for tile k+1 overlaps the
// 12 ds_load_b128 + 8 v_wmma of tile k; one barrier per K-step.
// ---------------------------------------------------------------------------
__global__ __launch_bounds__(256) void sru_gemm(const unsigned short* __restrict__ Xb,
                                                const unsigned short* __restrict__ Wt,
                                                float* __restrict__ U) {
    __shared__ __align__(16) unsigned short As[2][BM * ASTR];
    __shared__ __align__(16) unsigned short Bs[2][BN * ASTR];

    const int tid   = threadIdx.x;
    const int lane  = tid & 31;
    const int wave  = tid >> 5;
    const int waveM = wave >> 2;            // 0..1 -> 64 rows each
    const int waveN = wave & 3;             // 0..3 -> 32 cols each
    const int m0 = blockIdx.y * BM;
    const int n0 = blockIdx.x * BN;

    v8f acc[4][2];
#pragma unroll
    for (int i = 0; i < 4; ++i)
#pragma unroll
        for (int j = 0; j < 2; ++j)
            acc[i][j] = (v8f){0.f,0.f,0.f,0.f,0.f,0.f,0.f,0.f};

    const int am   = waveM * 64 + (lane & 15);
    const int kg   = (lane >> 4) * 8;    // A: lanes 0-15 K=0..7/16..23, 16-31 K=8..15/24..31
    const int bn   = waveN * 32 + (lane & 15);
    const int kb16 = (lane >> 4) * 8;    // B: lanes 0-15 K=0..7/16..23, 16-31 K=8..15/24..31

    // stage A (128 x 32 bf16) + B (128 x 32 bf16): 16B per lane-op
    auto stage = [&](unsigned short* Ad, unsigned short* Bd, int kb) {
#pragma unroll
        for (int t = 0; t < 2; ++t) {
            const int e   = t * 256 + tid;        // 0..511
            const int row = e >> 2;               // 0..127
            const int seg = (e & 3) * 8;          // u16 offset within row
            cp16(Ad + row * ASTR + seg, Xb + (size_t)(m0 + row) * KDIM + kb + seg);
            cp16(Bd + row * ASTR + seg, Wt + (size_t)(n0 + row) * KDIM + kb + seg);
        }
    };

    stage(As[0], Bs[0], 0);
    async_wait();
    __syncthreads();

    int buf = 0;
    for (int kb = 0; kb < KDIM; kb += TK) {
        // prefetch next tile into the alternate buffer (uniform branch)
        if (kb + TK < KDIM)
            stage(As[buf ^ 1], Bs[buf ^ 1], kb + TK);

        // ---- fragment loads from current buffer: all 16B ds_load_b128 ----
        Frag fa[4], fb[2];
#pragma unroll
        for (int i = 0; i < 4; ++i) {
            const unsigned short* p = As[buf] + (am + i * 16) * ASTR;
            fa[i].q[0] = *(const uint4*)(p + kg);
            fa[i].q[1] = *(const uint4*)(p + 16 + kg);
        }
#pragma unroll
        for (int j = 0; j < 2; ++j) {
            const unsigned short* p = Bs[buf] + (bn + j * 16) * ASTR + kb16;
            fb[j].q[0] = *(const uint4*)(p);
            fb[j].q[1] = *(const uint4*)(p + 16);
        }

#pragma unroll
        for (int i = 0; i < 4; ++i)
#pragma unroll
            for (int j = 0; j < 2; ++j)
                acc[i][j] = __builtin_amdgcn_wmma_f32_16x16x32_bf16(
                    false, fa[i].v, false, fb[j].v, (short)0, acc[i][j], false, false);

        // own async writes into next buffer done; barrier also guarantees all
        // waves finished reading the current buffer before it is re-staged.
        async_wait();
        __syncthreads();
        buf ^= 1;
    }

    // ---- epilogue: lanes 0-15 N=lane M=r, lanes 16-31 N=lane-16 M=r+8 ----
    const int moff = 8 * (lane >> 4);
    const int cn   = n0 + waveN * 32 + (lane & 15);
#pragma unroll
    for (int i = 0; i < 4; ++i)
#pragma unroll
        for (int j = 0; j < 2; ++j)
#pragma unroll
            for (int r = 0; r < 8; ++r) {
                const int mg = m0 + waveM * 64 + i * 16 + moff + r;
                U[(size_t)mg * NDIM + cn + j * 16] = acc[i][j][r];
            }
}

// ---------------------------------------------------------------------------
// Scan: one thread per (b, d); sequential over LCHUNK timesteps.
// u layout (L,B,3D) -> (L,B,D,3): comps at 3d, 3d+1, 3d+2 (contiguous 12 B).
// ---------------------------------------------------------------------------
__global__ __launch_bounds__(256) void sru_scan(const float* __restrict__ U,
                                                const float* __restrict__ X,
                                                const float* __restrict__ C0,
                                                const float* __restrict__ WC,
                                                const float* __restrict__ Bias,
                                                float* __restrict__ H,
                                                float* __restrict__ Cstate,
                                                float* __restrict__ Cfinal,
                                                int chunk) {
    const int idx = blockIdx.x * blockDim.x + threadIdx.x;  // 0..B*D-1
    const int d   = idx & (DDIM - 1);
    const int b   = idx >> 10;
    const float fw = WC[d],   rw = WC[DDIM + d];
    const float fb = Bias[d], rb = Bias[DDIM + d];
    const float SCALE_X = 1.7320508075688772f;              // sqrt(1 + 2*e^0)
    float c = (chunk == 0) ? C0[idx] : Cstate[idx];

    for (int l = 0; l < LCHUNK; ++l) {
        const size_t row = (size_t)l * BATCH + b;           // row within chunk
        const float* up = U + row * NDIM + 3 * d;
        const float u0 = up[0];
        const float u1 = up[1] + fb;
        const float u2 = up[2] + rb;
        const float xp = X[row * DDIM + d] * SCALE_X;
        const float f  = 1.0f / (1.0f + __expf(-(u1 + c * fw)));
        const float r  = 1.0f / (1.0f + __expf(-(u2 + c * rw)));
        c = u0 + (c - u0) * f;
        H[row * DDIM + d] = xp + (c - xp) * r;
    }
    Cstate[idx] = c;
    if (chunk == (L_TOTAL / LCHUNK) - 1) Cfinal[idx] = c;
}

extern "C" void kernel_launch(void* const* d_in, const int* in_sizes, int n_in,
                              void* d_out, int out_size, void* d_ws, size_t ws_size,
                              hipStream_t stream) {
    (void)in_sizes; (void)n_in; (void)out_size; (void)ws_size;
    const float* x    = (const float*)d_in[0];
    const float* c0   = (const float*)d_in[1];
    const float* w    = (const float*)d_in[2];
    const float* wc   = (const float*)d_in[3];
    const float* bias = (const float*)d_in[4];

    float* h    = (float*)d_out;
    float* cfin = (float*)d_out + (size_t)L_TOTAL * BATCH * DDIM;

    // workspace layout (all 16B aligned): U | cstate | Xb | Wt   (~62 MB)
    char* ws = (char*)d_ws;
    float*          u      = (float*)ws;                               // 48 MB
    ws += (size_t)MCHUNK * NDIM * sizeof(float);
    float*          cstate = (float*)ws;                               // 64 KB
    ws += (size_t)BATCH * DDIM * sizeof(float);
    unsigned short* xb     = (unsigned short*)ws;                      // 8 MB
    ws += (size_t)MCHUNK * KDIM * sizeof(unsigned short);
    unsigned short* wt     = (unsigned short*)ws;                      // 6 MB

    // one-time weight convert+transpose (Wt stays L2-resident: 6 MB of 192 MB)
    sru_wt<<<dim3(NDIM / 32, KDIM / 32), 256, 0, stream>>>(w, wt);

    const dim3 gemmGrid(NDIM / BN, MCHUNK / BM);    // (24, 32)
    const dim3 convGrid((MCHUNK * KDIM) / (256 * 8));
    const dim3 scanGrid((BATCH * DDIM) / 256);      // 64 blocks

    const int nchunks = L_TOTAL / LCHUNK;           // 8
    for (int ch = 0; ch < nchunks; ++ch) {
        const float* xch = x + (size_t)ch * LCHUNK * BATCH * DDIM;
        float*       hch = h + (size_t)ch * LCHUNK * BATCH * DDIM;
        sru_xb<<<convGrid, 256, 0, stream>>>(xch, xb);
        sru_gemm<<<gemmGrid, 256, 0, stream>>>(xb, wt, u);
        sru_scan<<<scanGrid, 256, 0, stream>>>(u, xch, c0, wc, bias,
                                               hch, cstate, cfin, ch);
    }
}
